// GCNBlock_65592740544800
// MI455X (gfx1250) — matrix operand order repaired and checked by reading, not compiled, not run on hardware
//
#include <hip/hip_runtime.h>

typedef __attribute__((ext_vector_type(16))) _Float16 v16h;
typedef __attribute__((ext_vector_type(8)))  _Float16 h8;
typedef __attribute__((ext_vector_type(8)))  float    v8f;

#define NN 8192
#define FF 64
#define BB 8
#define ROWS 32      // graph rows per workgroup (2 WMMA M-tiles)
#define KCHUNK 64    // K elements of adjacency staged to LDS per step

__device__ __forceinline__ void lds_fence() {
  asm volatile("s_wait_dscnt 0" ::: "memory");
}

// ---------------------------------------------------------------------------
// Prep: x[B][N][F] f32  ->  x_t[B][F][N] f16 (so WMMA B-fragments are
// contiguous per lane: column = feature, K = node index)
// ---------------------------------------------------------------------------
__global__ __launch_bounds__(256) void xt_transpose_kernel(
    const float* __restrict__ x, _Float16* __restrict__ xt) {
  __shared__ float tile[64][65];
  const int b  = blockIdx.y;
  const int n0 = blockIdx.x * 64;
  const int t  = threadIdx.x;
  {
    const int r = t >> 2;            // node row in tile (0..63)
    const int c = (t & 3) * 16;      // feature start
    const float4* src = (const float4*)(x + ((size_t)b * NN + n0 + r) * FF + c);
    float4 v0 = src[0], v1 = src[1], v2 = src[2], v3 = src[3];
    float* d = &tile[r][c];
    d[0]=v0.x; d[1]=v0.y; d[2]=v0.z; d[3]=v0.w;
    d[4]=v1.x; d[5]=v1.y; d[6]=v1.z; d[7]=v1.w;
    d[8]=v2.x; d[9]=v2.y; d[10]=v2.z; d[11]=v2.w;
    d[12]=v3.x; d[13]=v3.y; d[14]=v3.z; d[15]=v3.w;
  }
  __syncthreads();
  {
    const int f = t >> 2;            // feature (0..63)
    const int n = (t & 3) * 16;      // node start
    union { h8 v; _Float16 h[8]; } o0, o1;
    #pragma unroll
    for (int j = 0; j < 8; ++j) o0.h[j] = (_Float16)tile[n + j][f];
    #pragma unroll
    for (int j = 0; j < 8; ++j) o1.h[j] = (_Float16)tile[n + 8 + j][f];
    _Float16* dst = xt + ((size_t)b * FF + f) * NN + n0 + n;
    *(h8*)(dst)     = o0.v;
    *(h8*)(dst + 8) = o1.v;
  }
}

// ---------------------------------------------------------------------------
// Fused GraphSAGE: agg = A@x (WMMA f16, adj staged once through LDS),
// then agg/deg @ W_neigh, x @ W_self, bias + mask + relu + LayerNorm.
// Grid: N/ROWS = 256 blocks, 256 threads (8 waves).
// wave: mt = wave&1 selects 16-row M-tile, bg = wave>>1 selects batch pair.
// ---------------------------------------------------------------------------
__global__ __launch_bounds__(256) void sage_fused_kernel(
    const float* __restrict__ x, const int* __restrict__ adj,
    const float* __restrict__ Wself, const float* __restrict__ bself,
    const float* __restrict__ Wneigh, const float* __restrict__ bneigh,
    const float* __restrict__ gamma, const float* __restrict__ beta,
    const _Float16* __restrict__ xt, float* __restrict__ out) {
  __shared__ _Float16 A_s[ROWS][80];          // adj tile as f16 0/1
  __shared__ _Float16 Wt[2][64][80];          // [0]=W_self^T, [1]=W_neigh^T (f16)
  __shared__ float    degp[ROWS][8];
  __shared__ float    deg_s[ROWS];
  __shared__ float    gamma_s[64], beta_s[64], bs_s[64], bn_s[64];
  __shared__ float    epi[8][16][72];         // per-wave staging (agg / out rows)

  const int t    = threadIdx.x;
  const int lane = t & 31;
  const int wave = t >> 5;
  const int mt   = wave & 1;       // row tile 0/1
  const int bg   = wave >> 1;      // batch group 0..3 (batches 2bg, 2bg+1)
  const int i0   = blockIdx.x * ROWS;

  // ---- stage transposed weights (f16) + per-feature vectors ----
  {
    const int n  = t >> 2;         // out-feature 0..63
    const int k0 = (t & 3) * 16;   // in-feature start
    for (int kk = 0; kk < 16; ++kk) {
      Wt[0][n][k0 + kk] = (_Float16)Wself [(k0 + kk) * FF + n];
      Wt[1][n][k0 + kk] = (_Float16)Wneigh[(k0 + kk) * FF + n];
    }
    if (t < 64) {
      gamma_s[t] = gamma[t]; beta_s[t] = beta[t];
      bs_s[t]    = bself[t]; bn_s[t]   = bneigh[t];
    }
  }

  const v8f zero8 = {0.f,0.f,0.f,0.f,0.f,0.f,0.f,0.f};
  v8f acc[2][4];
  #pragma unroll
  for (int i = 0; i < 2; ++i)
    #pragma unroll
    for (int j = 0; j < 4; ++j) acc[i][j] = zero8;

  // per-lane fragment geometry
  const int mrow = lane & 15;
  const int koff = (lane < 16) ? 0 : 8;    // A-frag K interleave offset
  const int kbB  = (lane < 16) ? 0 : 8;    // B-frag K interleave offset

  const _Float16* xb[2][4];
  #pragma unroll
  for (int b2 = 0; b2 < 2; ++b2)
    #pragma unroll
    for (int ft = 0; ft < 4; ++ft) {
      const int b = bg * 2 + b2;
      const int f = ft * 16 + mrow;
      xb[b2][ft] = xt + ((size_t)b * FF + f) * NN + kbB;
    }

  // adjacency staging: thread t handles row ar, 8 consecutive k at ak
  const int ar = t >> 3;                    // 0..31
  const int ak = (t & 7) * 8;               // 0..56
  const int* arow = adj + (size_t)(i0 + ar) * NN + ak;
  int degi = 0;

  __syncthreads();

  // ---- main K loop: agg += A_tile @ x_tile -------------------------------
  for (int k0 = 0; k0 < NN; k0 += KCHUNK) {
    int4 a0 = *(const int4*)(arow + k0);
    int4 a1 = *(const int4*)(arow + k0 + 4);
    union { h8 v; _Float16 h[8]; } ah;
    const _Float16 one = (_Float16)1.0f, zer = (_Float16)0.0f;
    ah.h[0] = a0.x > 0 ? one : zer; degi += (a0.x > 0);
    ah.h[1] = a0.y > 0 ? one : zer; degi += (a0.y > 0);
    ah.h[2] = a0.z > 0 ? one : zer; degi += (a0.z > 0);
    ah.h[3] = a0.w > 0 ? one : zer; degi += (a0.w > 0);
    ah.h[4] = a1.x > 0 ? one : zer; degi += (a1.x > 0);
    ah.h[5] = a1.y > 0 ? one : zer; degi += (a1.y > 0);
    ah.h[6] = a1.z > 0 ? one : zer; degi += (a1.z > 0);
    ah.h[7] = a1.w > 0 ? one : zer; degi += (a1.w > 0);
    __syncthreads();                         // prior readers done
    *(h8*)&A_s[ar][ak] = ah.v;
    __syncthreads();                         // tile visible

    #pragma unroll
    for (int ks = 0; ks < 2; ++ks) {
      const int kk = ks * 32;
      union { v16h v; h8 h[2]; } af;
      af.h[0] = *(const h8*)&A_s[mt * 16 + mrow][kk + koff];
      af.h[1] = *(const h8*)&A_s[mt * 16 + mrow][kk + koff + 16];
      #pragma unroll
      for (int b2 = 0; b2 < 2; ++b2)
        #pragma unroll
        for (int ft = 0; ft < 4; ++ft) {
          union { v16h v; h8 h[2]; } bf;
          const _Float16* p = xb[b2][ft] + k0 + kk;
          bf.h[0] = *(const h8*)(p);
          bf.h[1] = *(const h8*)(p + 16);
          acc[b2][ft] = __builtin_amdgcn_wmma_f32_16x16x32_f16(
              false, af.v, false, bf.v, (short)0, acc[b2][ft], false, false);
        }
    }
  }

  // ---- degree reduction ---------------------------------------------------
  degp[ar][t & 7] = (float)degi;
  __syncthreads();
  if (t < ROWS) {
    float s = 0.f;
    #pragma unroll
    for (int j = 0; j < 8; ++j) s += degp[t][j];
    deg_s[t] = s;
  }
  __syncthreads();

  const int mrowsel = (lane < 16) ? 0 : 8;

  // ---- epilogue per batch owned by this wave ------------------------------
  for (int b2 = 0; b2 < 2; ++b2) {
    const int b = bg * 2 + b2;

    // 1) mean-scaled agg -> epi[wave] (row m, feature f)
    #pragma unroll
    for (int ft = 0; ft < 4; ++ft) {
      union { v8f v; float f[8]; } c; c.v = acc[b2][ft];
      #pragma unroll
      for (int r = 0; r < 8; ++r) {
        const int m = r + mrowsel;
        const float dg = deg_s[mt * 16 + m];
        const float sc = dg > 0.f ? 1.f / dg : 0.f;
        epi[wave][m][ft * 16 + mrow] = c.f[r] * sc;
      }
    }
    lds_fence();

    // 2)+3) neigh GEMM (epi @ Wt[1]) and self GEMM (x @ Wt[0]) via WMMA
    v8f accn[4], accs[4];
    #pragma unroll
    for (int j = 0; j < 4; ++j) { accn[j] = zero8; accs[j] = zero8; }

    #pragma unroll
    for (int ks = 0; ks < 2; ++ks) {
      const int kk = ks * 32;
      union { v16h v; _Float16 h[16]; } an, asf;
      {
        const float* src = &epi[wave][mrow][kk + koff];
        float4 u0 = *(const float4*)(src);
        float4 u1 = *(const float4*)(src + 4);
        float4 u2 = *(const float4*)(src + 16);
        float4 u3 = *(const float4*)(src + 20);
        an.h[0]=(_Float16)u0.x;  an.h[1]=(_Float16)u0.y;
        an.h[2]=(_Float16)u0.z;  an.h[3]=(_Float16)u0.w;
        an.h[4]=(_Float16)u1.x;  an.h[5]=(_Float16)u1.y;
        an.h[6]=(_Float16)u1.z;  an.h[7]=(_Float16)u1.w;
        an.h[8]=(_Float16)u2.x;  an.h[9]=(_Float16)u2.y;
        an.h[10]=(_Float16)u2.z; an.h[11]=(_Float16)u2.w;
        an.h[12]=(_Float16)u3.x; an.h[13]=(_Float16)u3.y;
        an.h[14]=(_Float16)u3.z; an.h[15]=(_Float16)u3.w;
      }
      {
        const float* sx = x + ((size_t)b * NN + i0 + mt * 16 + mrow) * FF + kk + koff;
        float4 u0 = *(const float4*)(sx);
        float4 u1 = *(const float4*)(sx + 4);
        float4 u2 = *(const float4*)(sx + 16);
        float4 u3 = *(const float4*)(sx + 20);
        asf.h[0]=(_Float16)u0.x;  asf.h[1]=(_Float16)u0.y;
        asf.h[2]=(_Float16)u0.z;  asf.h[3]=(_Float16)u0.w;
        asf.h[4]=(_Float16)u1.x;  asf.h[5]=(_Float16)u1.y;
        asf.h[6]=(_Float16)u1.z;  asf.h[7]=(_Float16)u1.w;
        asf.h[8]=(_Float16)u2.x;  asf.h[9]=(_Float16)u2.y;
        asf.h[10]=(_Float16)u2.z; asf.h[11]=(_Float16)u2.w;
        asf.h[12]=(_Float16)u3.x; asf.h[13]=(_Float16)u3.y;
        asf.h[14]=(_Float16)u3.z; asf.h[15]=(_Float16)u3.w;
      }
      #pragma unroll
      for (int ft = 0; ft < 4; ++ft) {
        union { v16h v; h8 h[2]; } bwN, bwS;
        const _Float16* pn = &Wt[1][ft * 16 + mrow][kk + kbB];
        bwN.h[0] = *(const h8*)(pn); bwN.h[1] = *(const h8*)(pn + 16);
        accn[ft] = __builtin_amdgcn_wmma_f32_16x16x32_f16(
            false, an.v, false, bwN.v, (short)0, accn[ft], false, false);
        const _Float16* ps = &Wt[0][ft * 16 + mrow][kk + kbB];
        bwS.h[0] = *(const h8*)(ps); bwS.h[1] = *(const h8*)(ps + 16);
        accs[ft] = __builtin_amdgcn_wmma_f32_16x16x32_f16(
            false, asf.v, false, bwS.v, (short)0, accs[ft], false, false);
      }
    }
    lds_fence();   // all epi reads done before overwrite

    // 4) bias + deg-mask + relu -> epi[wave] (row-major rows)
    #pragma unroll
    for (int ft = 0; ft < 4; ++ft) {
      union { v8f v; float f[8]; } cn, cs; cn.v = accn[ft]; cs.v = accs[ft];
      const int f = ft * 16 + mrow;
      const float bnv = bn_s[f], bsv = bs_s[f];
      #pragma unroll
      for (int r = 0; r < 8; ++r) {
        const int m = r + mrowsel;
        const float dg = deg_s[mt * 16 + m];
        float nv = cn.f[r] + bnv;  nv = dg > 0.f ? nv : 0.f;
        float v  = cs.f[r] + bsv + nv;
        epi[wave][m][f] = v > 0.f ? v : 0.f;
      }
    }
    lds_fence();

    // 5) LayerNorm + store (lanes 0..15 each own one row)
    if (lane < 16) {
      const int m = lane;
      float vals[64];
      #pragma unroll
      for (int q = 0; q < 16; ++q) {
        float4 u = *(const float4*)&epi[wave][m][q * 4];
        vals[q*4+0]=u.x; vals[q*4+1]=u.y; vals[q*4+2]=u.z; vals[q*4+3]=u.w;
      }
      float mu = 0.f;
      #pragma unroll
      for (int q = 0; q < 64; ++q) mu += vals[q];
      mu *= (1.f / 64.f);
      float var = 0.f;
      #pragma unroll
      for (int q = 0; q < 64; ++q) { float d = vals[q] - mu; var += d * d; }
      var *= (1.f / 64.f);
      const float rstd = __frsqrt_rn(var + 1e-5f);
      float* orow = out + ((size_t)b * NN + i0 + mt * 16 + m) * FF;
      #pragma unroll
      for (int q = 0; q < 16; ++q) {
        float4 w;
        w.x = (vals[q*4+0]-mu)*rstd*gamma_s[q*4+0] + beta_s[q*4+0];
        w.y = (vals[q*4+1]-mu)*rstd*gamma_s[q*4+1] + beta_s[q*4+1];
        w.z = (vals[q*4+2]-mu)*rstd*gamma_s[q*4+2] + beta_s[q*4+2];
        w.w = (vals[q*4+3]-mu)*rstd*gamma_s[q*4+3] + beta_s[q*4+3];
        *(float4*)(orow + q * 4) = w;
      }
    }
    lds_fence();   // readers done before next b2 overwrites epi
  }
}

// ---------------------------------------------------------------------------
extern "C" void kernel_launch(void* const* d_in, const int* in_sizes, int n_in,
                              void* d_out, int out_size, void* d_ws, size_t ws_size,
                              hipStream_t stream) {
  const float* x      = (const float*)d_in[0];
  const int*   adj    = (const int*)  d_in[1];
  const float* Wself  = (const float*)d_in[2];
  const float* bself  = (const float*)d_in[3];
  const float* Wneigh = (const float*)d_in[4];
  const float* bneigh = (const float*)d_in[5];
  const float* gamma  = (const float*)d_in[6];
  const float* beta   = (const float*)d_in[7];
  float*       out    = (float*)d_out;
  _Float16*    xt     = (_Float16*)d_ws;   // B*F*N f16 = 8 MB

  dim3 gprep(NN / 64, BB);
  xt_transpose_kernel<<<gprep, 256, 0, stream>>>(x, xt);

  sage_fused_kernel<<<NN / ROWS, 256, 0, stream>>>(
      x, adj, Wself, bself, Wneigh, bneigh, gamma, beta, xt, out);
}
